// Model_55104430408229
// MI455X (gfx1250) — matrix operand order repaired and checked by reading, not compiled, not run on hardware
//
#include <hip/hip_runtime.h>
#include <math.h>

#define Bdim   128
#define Ddim   128
#define Hdim   50
#define NEGn   4
#define Ldim   30
#define Sdim   55
#define NPGdim 80
#define Nn     10240
#define Ee     163840
#define NEWSn  7040
#define BEe    28160

typedef __attribute__((ext_vector_type(16))) _Float16 v16h;
typedef __attribute__((ext_vector_type(2)))  __fp16   v2fp16;
typedef __attribute__((ext_vector_type(8)))  float    v8f;

// ---------------- device helpers ----------------

__device__ __forceinline__ unsigned fkey(float f) {
    unsigned u = __float_as_uint(f);
    return (u & 0x80000000u) ? ~u : (u | 0x80000000u);
}
__device__ __forceinline__ float fkey_inv(unsigned k) {
    return (k & 0x80000000u) ? __uint_as_float(k & 0x7FFFFFFFu) : __uint_as_float(~k);
}
// pack two f32 -> packed f16 pair (v_cvt_pk_f16_f32 / pkrtz HW pack op)
__device__ __forceinline__ unsigned pk2(float a, float b) {
    v2fp16 h = __builtin_amdgcn_cvt_pkrtz(a, b);
    return __builtin_bit_cast(unsigned, h);
}

// ---------------- WMMA GEMM: C[M x 128] = A[M x 128] @ W[128 x 128] ----------------
// block = 128 threads = 4 waves; each wave owns one 16x16 C tile; block covers 64 rows x 16 cols.
// LDS tiles are stored as packed f16 in 128-bit quanta so staging is ds_store_b128 and
// fragment builds are ds_load_b128 (2 per operand per wave per K-step).
__global__ __launch_bounds__(128) void gemm_wmma(const float* __restrict__ Ag,
                                                 const float* __restrict__ Wg,
                                                 float* __restrict__ Cg, int M) {
    (void)M;
    __shared__ uint4 As[64][4];   // 64 rows x 32 packed halves (K-contiguous)
    __shared__ uint4 Bs[16][4];   // 16 cols x 32 packed halves (K-contiguous per column)
    const int tid   = threadIdx.x;
    const int wave  = tid >> 5;
    const int lane  = tid & 31;
    const int m_blk = blockIdx.y * 64;
    const int n0    = blockIdx.x * 16;
    const int hi    = lane >> 4;
    const int mrow  = (lane & 15) + wave * 16;
    const int ncol  = lane & 15;
    v8f acc = {};
    for (int k0 = 0; k0 < 128; k0 += 32) {
        // stage A tile: 64x32 f32 -> f16; 8 floats (2x float4) per thread per iter
        #pragma unroll
        for (int it = 0; it < 2; ++it) {
            int idx = it * 128 + tid;          // 0..255
            int r = idx >> 2, q = idx & 3;     // row, 8-half slot
            const float4* src = (const float4*)(Ag + (size_t)(m_blk + r) * 128 + k0 + 8 * q);
            float4 f0 = src[0], f1 = src[1];
            uint4 p;
            p.x = pk2(f0.x, f0.y); p.y = pk2(f0.z, f0.w);
            p.z = pk2(f1.x, f1.y); p.w = pk2(f1.z, f1.w);
            As[r][q] = p;
        }
        // stage B tile transposed: Bs[c][q] holds K halves [8q,8q+8) of output column n0+c
        if (tid < 64) {
            int c = tid >> 2, q = tid & 3;
            float f0 = Wg[(size_t)(k0 + 8 * q + 0) * 128 + n0 + c];
            float f1 = Wg[(size_t)(k0 + 8 * q + 1) * 128 + n0 + c];
            float f2 = Wg[(size_t)(k0 + 8 * q + 2) * 128 + n0 + c];
            float f3 = Wg[(size_t)(k0 + 8 * q + 3) * 128 + n0 + c];
            float f4 = Wg[(size_t)(k0 + 8 * q + 4) * 128 + n0 + c];
            float f5 = Wg[(size_t)(k0 + 8 * q + 5) * 128 + n0 + c];
            float f6 = Wg[(size_t)(k0 + 8 * q + 6) * 128 + n0 + c];
            float f7 = Wg[(size_t)(k0 + 8 * q + 7) * 128 + n0 + c];
            uint4 p;
            p.x = pk2(f0, f1); p.y = pk2(f2, f3);
            p.z = pk2(f4, f5); p.w = pk2(f6, f7);
            Bs[c][q] = p;
        }
        __syncthreads();
        // fragment build: A lane needs K {base..base+7} U {16+base..16+base+7}, base = 8*hi
        // -> two contiguous 16B chunks. B (K-major per column) likewise: same K set as A.
        union { uint4 u[2]; v16h h; } ua, ub;
        ua.u[0] = As[mrow][hi];
        ua.u[1] = As[mrow][2 + hi];
        ub.u[0] = Bs[ncol][hi];
        ub.u[1] = Bs[ncol][2 + hi];
        acc = __builtin_amdgcn_wmma_f32_16x16x32_f16(false, ua.h, false, ub.h,
                                                     (short)0, acc, false, false);
        __syncthreads();
    }
    const int mbase = m_blk + wave * 16 + (hi ? 8 : 0);
    #pragma unroll
    for (int vg = 0; vg < 8; ++vg)
        Cg[(size_t)(mbase + vg) * 128 + n0 + ncol] = acc[vg];
}

// ---------------- utility kernels ----------------

__global__ void k_fill_zero(float* p, long n) {
    long i = (long)blockIdx.x * blockDim.x + threadIdx.x;
    if (i < n) p[i] = 0.0f;
}

__global__ void k_gather_rows(const float* __restrict__ tab, const int* __restrict__ idx,
                              float* __restrict__ out, int rows) {
    int r = blockIdx.x;
    if (r >= rows) return;
    out[(size_t)r * 128 + threadIdx.x] = tab[(size_t)idx[r] * 128 + threadIdx.x];
}

// warp-per-row dot of a 128-vector with a shared vector (+optional scalar bias)
__global__ __launch_bounds__(128) void k_dot_vec(const float* __restrict__ h,
                                                 const float* __restrict__ a,
                                                 const float* __restrict__ bias,
                                                 float* __restrict__ out, int n) {
    int r = blockIdx.x * 4 + (threadIdx.x >> 5);
    if (r >= n) return;
    int lane = threadIdx.x & 31;
    float p = 0.f;
    #pragma unroll
    for (int j = 0; j < 4; ++j) {
        int d = lane + 32 * j;
        p += h[(size_t)r * 128 + d] * a[d];
    }
    #pragma unroll
    for (int off = 16; off > 0; off >>= 1) p += __shfl_xor(p, off, 32);
    if (lane == 0) out[r] = p + (bias ? bias[0] : 0.f);
}

// ---------------- title encoder ----------------
// one block (128 threads) per title row t = b*S + s
__global__ __launch_bounds__(128) void k_encode_titles(
    const int* __restrict__ hist_seqs, const int* __restrict__ hist_lens,
    const int* __restrict__ pos_seq,   const int* __restrict__ pos_len,
    const int* __restrict__ neg_seqs,  const int* __restrict__ neg_lens,
    const int* __restrict__ title_emb, const float* __restrict__ word_emb,
    const float* __restrict__ teW, const float* __restrict__ teb,
    const float* __restrict__ tev, float* __restrict__ out) {
    int t = blockIdx.x;
    int b = t / Sdim, s = t % Sdim;
    int tid = threadIdx.x;
    int nid, len;
    if (s < Hdim)      { nid = hist_seqs[b * Hdim + s]; len = hist_lens[b * Hdim + s]; }
    else if (s == Hdim){ nid = pos_seq[b];              len = pos_len[b]; }
    else               { int j = s - Hdim - 1; nid = neg_seqs[b * NEGn + j]; len = neg_lens[b * NEGn + j]; }

    __shared__ float embs[Ldim * 128];
    __shared__ float sc[32];
    __shared__ float red[128];
    for (int l = 0; l < Ldim; ++l) {
        int tok = title_emb[(size_t)nid * Ldim + l];
        embs[l * 128 + tid] = word_emb[(size_t)tok * 128 + tid];
    }
    __syncthreads();
    for (int l = 0; l < Ldim; ++l) {
        float hj = teb[tid];
        for (int d = 0; d < 128; ++d) hj += embs[l * 128 + d] * teW[d * 128 + tid];
        red[tid] = tanhf(hj) * tev[tid];
        __syncthreads();
        for (int ss = 64; ss > 0; ss >>= 1) { if (tid < ss) red[tid] += red[tid + ss]; __syncthreads(); }
        if (tid == 0) sc[l] = red[0];
        __syncthreads();
    }
    if (tid == 0) {
        float mx = -1e30f;
        for (int l = 0; l < Ldim; ++l) { float v = (l < len) ? sc[l] : -1e9f; sc[l] = v; mx = fmaxf(mx, v); }
        float dn = 0.f;
        for (int l = 0; l < Ldim; ++l) { float e = expf(sc[l] - mx); sc[l] = e; dn += e; }
        for (int l = 0; l < Ldim; ++l) sc[l] /= dn;
    }
    __syncthreads();
    float o = 0.f;
    for (int l = 0; l < Ldim; ++l) o += sc[l] * embs[l * 128 + tid];
    out[(size_t)t * 128 + tid] = o;
}

// ---------------- self attention (block per (b, query i)) ----------------
__global__ __launch_bounds__(128) void k_self_attn(const float* __restrict__ Q,
                                                   const float* __restrict__ K,
                                                   const float* __restrict__ V,
                                                   float* __restrict__ out, int outStride) {
    int t = blockIdx.x;
    int b = t / Hdim, i = t % Hdim;
    int tid = threadIdx.x;
    __shared__ float qs[128];
    __shared__ float sc[64];
    qs[tid] = Q[((size_t)b * Sdim + i) * 128 + tid];
    __syncthreads();
    if (tid < Hdim) {
        const float* kr = K + ((size_t)b * Sdim + tid) * 128;
        float s = 0.f;
        for (int d = 0; d < 128; ++d) s += qs[d] * kr[d];
        sc[tid] = s * 0.08838834764831845f;  // 1/sqrt(128)
    }
    __syncthreads();
    if (tid == 0) {
        float mx = -1e30f;
        for (int j = 0; j < Hdim; ++j) mx = fmaxf(mx, sc[j]);
        float dn = 0.f;
        for (int j = 0; j < Hdim; ++j) { float e = expf(sc[j] - mx); sc[j] = e; dn += e; }
        for (int j = 0; j < Hdim; ++j) sc[j] /= dn;
    }
    __syncthreads();
    float o = 0.f;
    for (int j = 0; j < Hdim; ++j) o += sc[j] * V[((size_t)b * Sdim + j) * 128 + tid];
    out[((size_t)b * outStride + i) * 128 + tid] = o;
}

// ---------------- edge pipeline (segment softmax + aggregate) ----------------

__global__ void k_edge_score(const int* __restrict__ src, const int* __restrict__ dst,
                             const float* __restrict__ ssrc, const float* __restrict__ sdst,
                             float* __restrict__ e, unsigned* __restrict__ mkey, int nE) {
    int i = blockIdx.x * 256 + threadIdx.x;
    if (i >= nE) return;
    float v = ssrc[src[i]] + sdst[dst[i]];
    v = v > 0.f ? v : 0.2f * v;  // leaky relu
    e[i] = v;
    atomicMax(&mkey[dst[i]], fkey(v));
}

__global__ void k_edge_expsum(const int* __restrict__ dst, const float* __restrict__ e,
                              const unsigned* __restrict__ mkey, float* __restrict__ ex,
                              float* __restrict__ den, int nE) {
    int i = blockIdx.x * 256 + threadIdx.x;
    if (i >= nE) return;
    float m = fkey_inv(mkey[dst[i]]);
    float ev = expf(e[i] - m);
    ex[i] = ev;
    atomicAdd(&den[dst[i]], ev);
}

// warp per edge: agg[dst] += alpha * h[src]
__global__ __launch_bounds__(128) void k_edge_agg(const int* __restrict__ src,
                                                  const int* __restrict__ dst,
                                                  const float* __restrict__ ex,
                                                  const float* __restrict__ den,
                                                  const float* __restrict__ hsrc,
                                                  float* __restrict__ agg, int nE) {
    int e = blockIdx.x * 4 + (threadIdx.x >> 5);
    if (e >= nE) return;
    int lane = threadIdx.x & 31;
    int s = src[e], dgt = dst[e];
    float alpha = ex[e] / (den[dgt] + 1e-16f);
    #pragma unroll
    for (int j = 0; j < 4; ++j) {
        int d = lane + 32 * j;
        atomicAdd(&agg[(size_t)dgt * 128 + d], alpha * hsrc[(size_t)s * 128 + d]);
    }
}

__global__ void k_finalize(const float* __restrict__ agg, const float* __restrict__ resid,
                           float* __restrict__ out, long n) {
    long i = (long)blockIdx.x * blockDim.x + threadIdx.x;
    if (i >= n) return;
    float v = agg[i];
    v = v > 0.f ? v : expm1f(v);  // elu
    if (resid) v += resid[i];
    out[i] = v;
}

// ---------------- attend pool over (B, H, D) with all-ones mask ----------------
__global__ __launch_bounds__(128) void k_attend_pool(const float* __restrict__ emb,
                                                     const float* __restrict__ Wm,
                                                     const float* __restrict__ bias,
                                                     const float* __restrict__ vvec,
                                                     float* __restrict__ out) {
    int b = blockIdx.x, tid = threadIdx.x;
    __shared__ float sc[64];
    __shared__ float red[128];
    for (int l = 0; l < Hdim; ++l) {
        const float* er = emb + ((size_t)b * Hdim + l) * 128;
        float hj = bias[tid];
        for (int d = 0; d < 128; ++d) hj += er[d] * Wm[d * 128 + tid];
        red[tid] = tanhf(hj) * vvec[tid];
        __syncthreads();
        for (int s = 64; s > 0; s >>= 1) { if (tid < s) red[tid] += red[tid + s]; __syncthreads(); }
        if (tid == 0) sc[l] = red[0];
        __syncthreads();
    }
    if (tid == 0) {
        float mx = -1e30f;
        for (int l = 0; l < Hdim; ++l) mx = fmaxf(mx, sc[l]);
        float dn = 0.f;
        for (int l = 0; l < Hdim; ++l) { float e = expf(sc[l] - mx); sc[l] = e; dn += e; }
        for (int l = 0; l < Hdim; ++l) sc[l] /= dn;
    }
    __syncthreads();
    float o = 0.f;
    for (int l = 0; l < Hdim; ++l) o += sc[l] * emb[((size_t)b * Hdim + l) * 128 + tid];
    out[(size_t)b * 128 + tid] = o;
}

// ---------------- mask pool: block per (b, slot) over NPG=80 contiguous nodes ----------------
__global__ __launch_bounds__(128) void k_mask_pool(const float* __restrict__ x,
                                                   const float* __restrict__ gate_s,
                                                   const int* __restrict__ hist_mask,
                                                   const int* __restrict__ pos_mask,
                                                   const int* __restrict__ neg_masks,
                                                   float* __restrict__ pooled) {
    int blk = blockIdx.x;
    int b = blk / 6, slot = blk % 6;
    int tid = threadIdx.x;
    __shared__ float w[NPGdim];
    __shared__ int   mf[NPGdim];
    if (tid < NPGdim) {
        int node = b * NPGdim + tid;
        int mv;
        if (slot == 0)      mv = hist_mask[node];
        else if (slot == 1) mv = pos_mask[node];
        else                mv = neg_masks[(size_t)node * NEGn + (slot - 2)];
        mf[tid] = (mv > 0);
        w[tid] = (mv > 0) ? gate_s[node] : -1e9f;
    }
    __syncthreads();
    if (tid == 0) {
        float mx = -1e30f;
        for (int i = 0; i < NPGdim; ++i) mx = fmaxf(mx, w[i]);
        float dn = 0.f;
        for (int i = 0; i < NPGdim; ++i) { float e = mf[i] ? expf(w[i] - mx) : 0.f; w[i] = e; dn += e; }
        float inv = 1.f / (dn + 1e-16f);
        for (int i = 0; i < NPGdim; ++i) w[i] *= inv;
    }
    __syncthreads();
    float o = 0.f;
    for (int i = 0; i < NPGdim; ++i) o += w[i] * x[((size_t)b * NPGdim + i) * 128 + tid];
    pooled[((size_t)b * 6 + slot) * 128 + tid] = o;
}

// ---------------- final gated fusion + dot score: block per (b, k) ----------------
__global__ __launch_bounds__(128) void k_final(const float* __restrict__ pooled,
                                               const float* __restrict__ uservec,
                                               const float* __restrict__ news,
                                               const float* __restrict__ wwW,
                                               const float* __restrict__ wwb,
                                               float* __restrict__ out) {
    int blk = blockIdx.x;
    int b = blk / (NEGn + 1), k = blk % (NEGn + 1);
    int tid = threadIdx.x;
    __shared__ float red[128];
    __shared__ float sh[2];
    float g_u = pooled[((size_t)b * 6 + 0) * 128 + tid];
    float t_u = uservec[(size_t)b * 128 + tid];
    float g_t = pooled[((size_t)b * 6 + 1 + k) * 128 + tid];
    float t_t = news[((size_t)b * Sdim + Hdim + k) * 128 + tid];
    float bb = wwb[0];
    red[tid] = g_u * wwW[tid] + t_u * wwW[128 + tid];
    __syncthreads();
    for (int s = 64; s > 0; s >>= 1) { if (tid < s) red[tid] += red[tid + s]; __syncthreads(); }
    if (tid == 0) sh[0] = 1.f / (1.f + expf(-(red[0] + bb)));
    __syncthreads();
    red[tid] = g_t * wwW[tid] + t_t * wwW[128 + tid];
    __syncthreads();
    for (int s = 64; s > 0; s >>= 1) { if (tid < s) red[tid] += red[tid + s]; __syncthreads(); }
    if (tid == 0) sh[1] = 1.f / (1.f + expf(-(red[0] + bb)));
    __syncthreads();
    float uw = sh[0], tw = sh[1];
    float uh = uw * g_u + (1.f - uw) * t_u;
    float th = tw * g_t + (1.f - tw) * t_t;
    red[tid] = uh * th;
    __syncthreads();
    for (int s = 64; s > 0; s >>= 1) { if (tid < s) red[tid] += red[tid + s]; __syncthreads(); }
    if (tid == 0) out[blk] = red[0];
}

// ---------------- host helpers ----------------

static void run_gemm(const float* A, const float* W, float* C, int M, hipStream_t st) {
    dim3 g(Ddim / 16, M / 64);
    gemm_wmma<<<g, 128, 0, st>>>(A, W, C, M);
}

static void run_seg(const int* src, const int* dst, int nE, int nDst,
                    const float* ssrc, const float* sdst, const float* hsrc, float* agg,
                    float* ebuf, float* exbuf, unsigned* mkey, float* den, hipStream_t st) {
    long aggN = (long)nDst * 128;
    k_fill_zero<<<(int)((aggN + 255) / 256), 256, 0, st>>>(agg, aggN);
    k_fill_zero<<<(nDst + 255) / 256, 256, 0, st>>>((float*)mkey, nDst);
    k_fill_zero<<<(nDst + 255) / 256, 256, 0, st>>>(den, nDst);
    k_edge_score<<<(nE + 255) / 256, 256, 0, st>>>(src, dst, ssrc, sdst, ebuf, mkey, nE);
    k_edge_expsum<<<(nE + 255) / 256, 256, 0, st>>>(dst, ebuf, mkey, exbuf, den, nE);
    k_edge_agg<<<(nE + 3) / 4, 128, 0, st>>>(src, dst, exbuf, den, hsrc, agg, nE);
}

static void run_gat(const float* x, const float* W, const float* as_, const float* ad_,
                    const int* src, const int* dst, float* outp,
                    float* proj, float* ssrc, float* sdst, float* agg,
                    float* ebuf, float* exbuf, unsigned* mkey, float* den, hipStream_t st) {
    run_gemm(x, W, proj, Nn, st);
    k_dot_vec<<<(Nn + 3) / 4, 128, 0, st>>>(proj, as_, nullptr, ssrc, Nn);
    k_dot_vec<<<(Nn + 3) / 4, 128, 0, st>>>(proj, ad_, nullptr, sdst, Nn);
    run_seg(src, dst, Ee, Nn, ssrc, sdst, proj, agg, ebuf, exbuf, mkey, den, st);
    long n = (long)Nn * 128;
    k_finalize<<<(int)((n + 255) / 256), 256, 0, st>>>(agg, nullptr, outp, n);
}

static void run_cross(const float* news_in, const float* node_in,
                      const float* Ws, const float* Wd, const float* as_, const float* ad_,
                      const int* bi_news, const int* bi_node,
                      float* node_out, float* news_out,
                      float* hsN, float* hdN, float* hsV, float* hdV,
                      float* ssrc, float* sdst, float* agg,
                      float* ebuf, float* exbuf, unsigned* mkey, float* den, hipStream_t st) {
    run_gemm(news_in, Ws, hsN, NEWSn, st);
    run_gemm(news_in, Wd, hdN, NEWSn, st);
    run_gemm(node_in, Ws, hsV, Nn, st);
    run_gemm(node_in, Wd, hdV, Nn, st);
    // direction 1: news -> node
    k_dot_vec<<<(NEWSn + 3) / 4, 128, 0, st>>>(hsN, as_, nullptr, ssrc, NEWSn);
    k_dot_vec<<<(Nn + 3) / 4, 128, 0, st>>>(hdV, ad_, nullptr, sdst, Nn);
    run_seg(bi_news, bi_node, BEe, Nn, ssrc, sdst, hsN, agg, ebuf, exbuf, mkey, den, st);
    long n1 = (long)Nn * 128;
    k_finalize<<<(int)((n1 + 255) / 256), 256, 0, st>>>(agg, node_in, node_out, n1);
    // direction 2: node -> news
    k_dot_vec<<<(Nn + 3) / 4, 128, 0, st>>>(hsV, as_, nullptr, ssrc, Nn);
    k_dot_vec<<<(NEWSn + 3) / 4, 128, 0, st>>>(hdN, ad_, nullptr, sdst, NEWSn);
    run_seg(bi_node, bi_news, BEe, NEWSn, ssrc, sdst, hsV, agg, ebuf, exbuf, mkey, den, st);
    long n2 = (long)NEWSn * 128;
    k_finalize<<<(int)((n2 + 255) / 256), 256, 0, st>>>(agg, news_in, news_out, n2);
}

static void run_selfattn(const float* in, const float* Wq, const float* Wk, const float* Wv,
                         float* Qb, float* Kb, float* Vb, float* outp, int outStride,
                         hipStream_t st) {
    run_gemm(in, Wq, Qb, NEWSn, st);
    run_gemm(in, Wk, Kb, NEWSn, st);
    run_gemm(in, Wv, Vb, NEWSn, st);
    k_self_attn<<<Bdim * Hdim, 128, 0, st>>>(Qb, Kb, Vb, outp, outStride);
}

// ---------------- entry ----------------

extern "C" void kernel_launch(void* const* d_in, const int* in_sizes, int n_in,
                              void* d_out, int out_size, void* d_ws, size_t ws_size,
                              hipStream_t stream) {
    (void)in_sizes; (void)n_in; (void)out_size; (void)ws_size;
    const int* nodes     = (const int*)d_in[0];
    const int* edge      = (const int*)d_in[1];
    const int* esrc      = edge;
    const int* edst      = edge + Ee;
    const int* bi        = (const int*)d_in[2];
    const int* bi_news   = bi;
    const int* bi_node   = bi + BEe;
    const int* hist_mask = (const int*)d_in[4];
    const int* pos_mask  = (const int*)d_in[5];
    const int* neg_masks = (const int*)d_in[6];
    const int* hist_seqs = (const int*)d_in[7];
    const int* hist_lens = (const int*)d_in[8];
    const int* pos_seq   = (const int*)d_in[9];
    const int* pos_len   = (const int*)d_in[10];
    const int* neg_seqs  = (const int*)d_in[11];
    const int* neg_lens  = (const int*)d_in[12];
    const int* title_emb = (const int*)d_in[13];
    const float* ent_emb = (const float*)d_in[14];
    const float* word_emb= (const float*)d_in[15];
    const float* teW = (const float*)d_in[16];
    const float* teb = (const float*)d_in[17];
    const float* tev = (const float*)d_in[18];
    const float* ue1q = (const float*)d_in[19], *ue1k = (const float*)d_in[20], *ue1v = (const float*)d_in[21];
    const float* ue2q = (const float*)d_in[22], *ue2k = (const float*)d_in[23], *ue2v = (const float*)d_in[24];
    const float* ue3q = (const float*)d_in[25], *ue3k = (const float*)d_in[26], *ue3v = (const float*)d_in[27];
    const float* g1W = (const float*)d_in[28], *g1as = (const float*)d_in[29], *g1ad = (const float*)d_in[30];
    const float* g2W = (const float*)d_in[31], *g2as = (const float*)d_in[32], *g2ad = (const float*)d_in[33];
    const float* cg1Ws = (const float*)d_in[34], *cg1Wd = (const float*)d_in[35];
    const float* cg1as = (const float*)d_in[36], *cg1ad = (const float*)d_in[37];
    const float* cg2Ws = (const float*)d_in[38], *cg2Wd = (const float*)d_in[39];
    const float* cg2as = (const float*)d_in[40], *cg2ad = (const float*)d_in[41];
    const float* saW = (const float*)d_in[42], *sab = (const float*)d_in[43], *sav = (const float*)d_in[44];
    const float* gateW = (const float*)d_in[45], *gateb = (const float*)d_in[46];
    const float* wwW = (const float*)d_in[47], *wwb = (const float*)d_in[48];

    // workspace arena
    float* ws = (float*)d_ws;
    size_t off = 0;
    auto A = [&](size_t n) { float* p = ws + off; off += n; return p; };
    float* titles0 = A((size_t)NEWSn * 128);
    float* news_a  = A((size_t)NEWSn * 128);
    float* news_b  = A((size_t)NEWSn * 128);
    float* node_a  = A((size_t)Nn * 128);
    float* node_b  = A((size_t)Nn * 128);
    float* projV1  = A((size_t)Nn * 128);
    float* projV2  = A((size_t)Nn * 128);
    float* projN1  = A((size_t)NEWSn * 128);
    float* projN2  = A((size_t)NEWSn * 128);
    float* Qb      = A((size_t)NEWSn * 128);
    float* Kb      = A((size_t)NEWSn * 128);
    float* Vb      = A((size_t)NEWSn * 128);
    float* agg     = A((size_t)Nn * 128);
    float* ebuf    = A(Ee);
    float* exbuf   = A(Ee);
    unsigned* mkey = (unsigned*)A(Nn);
    float* den     = A(Nn);
    float* ssrc    = A(Nn);
    float* sdst    = A(Nn);
    float* gates   = A(Nn);
    float* ut3     = A((size_t)Bdim * Hdim * 128);
    float* uservec = A((size_t)Bdim * 128);
    float* pooled  = A((size_t)Bdim * 6 * 128);

    const size_t newsBytes = (size_t)NEWSn * 128 * sizeof(float);

    // Stage 0: encode titles into (B,S,D) layout; gather entity embeddings
    k_encode_titles<<<NEWSn, 128, 0, stream>>>(hist_seqs, hist_lens, pos_seq, pos_len,
                                               neg_seqs, neg_lens, title_emb, word_emb,
                                               teW, teb, tev, titles0);
    k_gather_rows<<<Nn, 128, 0, stream>>>(ent_emb, nodes, node_a, Nn);

    // Stage 1: news1 = [self_attn(user titles, ue1), target titles]
    (void)hipMemcpyAsync(news_a, titles0, newsBytes, hipMemcpyDeviceToDevice, stream);
    run_selfattn(titles0, ue1q, ue1k, ue1v, Qb, Kb, Vb, news_a, Sdim, stream);

    // Stage 2: node1 = gat(node0, g1)        node_a -> node_b
    run_gat(node_a, g1W, g1as, g1ad, esrc, edst, node_b,
            projV1, ssrc, sdst, agg, ebuf, exbuf, mkey, den, stream);

    // Stage 3: cross1(news1, node1) -> node_cross1 (node_a), news_cross1 (news_b)
    run_cross(news_a, node_b, cg1Ws, cg1Wd, cg1as, cg1ad, bi_news, bi_node,
              node_a, news_b, projN1, projN2, projV1, projV2,
              ssrc, sdst, agg, ebuf, exbuf, mkey, den, stream);

    // Stage 4: news2 = [self_attn(user rows of news_cross1, ue2), targets]  -> news_a
    (void)hipMemcpyAsync(news_a, news_b, newsBytes, hipMemcpyDeviceToDevice, stream);
    run_selfattn(news_b, ue2q, ue2k, ue2v, Qb, Kb, Vb, news_a, Sdim, stream);

    // Stage 5: node2 = gat(node_cross1, g2)   node_a -> node_b
    run_gat(node_a, g2W, g2as, g2ad, esrc, edst, node_b,
            projV1, ssrc, sdst, agg, ebuf, exbuf, mkey, den, stream);

    // Stage 6: cross2(news2, node2) -> node_h (node_a), news_cross2 (news_b)
    run_cross(news_a, node_b, cg2Ws, cg2Wd, cg2as, cg2ad, bi_news, bi_node,
              node_a, news_b, projN1, projN2, projV1, projV2,
              ssrc, sdst, agg, ebuf, exbuf, mkey, den, stream);

    // Stage 7: user_title3 = self_attn(user rows of news_cross2, ue3) -> ut3 (B,H,D)
    run_selfattn(news_b, ue3q, ue3k, ue3v, Qb, Kb, Vb, ut3, Hdim, stream);
    k_attend_pool<<<Bdim, 128, 0, stream>>>(ut3, saW, sab, sav, uservec);

    // Stage 8: gate + mask pool over node_h
    k_dot_vec<<<(Nn + 3) / 4, 128, 0, stream>>>(node_a, gateW, gateb, gates, Nn);
    k_mask_pool<<<Bdim * 6, 128, 0, stream>>>(node_a, gates, hist_mask, pos_mask, neg_masks, pooled);

    // Stage 9: final gated fusion + scores -> (B, NEG+1)
    k_final<<<Bdim * (NEGn + 1), 128, 0, stream>>>(pooled, uservec, news_b, wwW, wwb, (float*)d_out);
}
